// CoAttention_79405355368478
// MI455X (gfx1250) — matrix operand order repaired and checked by reading, not compiled, not run on hardware
//
#include <hip/hip_runtime.h>

#define DDIM 256
#define ODIM 128
#define NEGV -1e18f

typedef __attribute__((ext_vector_type(16))) __bf16       v16bf;
typedef __attribute__((ext_vector_type(8)))  float        v8f;
typedef __attribute__((ext_vector_type(8)))  unsigned int v8u;

__device__ __forceinline__ unsigned int pack2bf(float lo, float hi) {
    // truncating f32 -> bf16, packed {hi, lo}
    return (__float_as_uint(hi) & 0xFFFF0000u) | (__float_as_uint(lo) >> 16);
}

// Hardware tanh on gfx1250 (V_TANH_F32) if the builtin is exposed; cheap
// branch-free rational fallback otherwise (tanh saturates well before |x|=15).
#if __has_builtin(__builtin_amdgcn_tanhf)
__device__ __forceinline__ float fast_tanh(float x) { return __builtin_amdgcn_tanhf(x); }
#elif __has_builtin(__builtin_amdgcn_tanh_f32)
__device__ __forceinline__ float fast_tanh(float x) { return __builtin_amdgcn_tanh_f32(x); }
#else
__device__ __forceinline__ float fast_tanh(float x) {
    const float xc = fminf(fmaxf(x, -15.f), 15.f);
    const float t  = __expf(2.f * xc);
    return __fdividef(t - 1.f, t + 1.f);
}
#endif

// One workgroup per (b,e,side). Blocks 0..255: evidence side (L=512);
// blocks 256..511: claim side (L=256). 256 threads = 8 waves.
// h = tanh(seq @ W[:D] + pooled@W[D:]); att = h@w; masked softmax over L;
// out = scores^T @ seq.
__global__ __launch_bounds__(256)
void coattend_kernel(const float* __restrict__ claim, const int* __restrict__ cmask,
                     const float* __restrict__ evid,  const int* __restrict__ emask,
                     const float* __restrict__ W1, const float* __restrict__ w2,
                     const float* __restrict__ W2, const float* __restrict__ w1,
                     float* __restrict__ c_hat, float* __restrict__ e_hat)
{
    __shared__ __align__(16) unsigned short sW[ODIM * DDIM]; // bf16 W[:D]^T, [n][k], 64KB
    __shared__ float sAtt[512];        // logits -> exp values (max L)
    __shared__ float sVec[256];        // pooled mean, then reduction scratch
    __shared__ float sPbias[ODIM];
    __shared__ float sw[ODIM];

    const int  tid      = threadIdx.x;
    const int  blk      = blockIdx.x;
    const bool evidSide = blk < 256;
    const int  pair     = evidSide ? blk : (blk - 256);
    const int  L        = evidSide ? 512 : 256;
    const int  LOTHER   = evidSide ? 256 : 512;

    const float* seq   = (evidSide ? evid  : claim) + (size_t)pair * L * DDIM;
    const int*   mask  = (evidSide ? emask : cmask) + pair * L;
    const float* other = (evidSide ? claim : evid)  + (size_t)pair * LOTHER * DDIM;
    const float* W     = evidSide ? W1 : W2;     // (2D, OUT) row-major
    const float* wv    = evidSide ? w2 : w1;     // (OUT,)
    float*       out   = (evidSide ? e_hat : c_hat) + pair * DDIM;

    // --- stage W[:D] transposed into LDS as bf16 (col-major by n) ---
    for (int idx = tid; idx < DDIM * ODIM; idx += 256) {
        const int k = idx >> 7;          // row of W (K dim)
        const int n = idx & (ODIM - 1);  // col of W (N dim)
        sW[n * DDIM + k] = (unsigned short)(__float_as_uint(W[idx]) >> 16);
    }
    // --- pooled mean of the other sequence (thread t = dim t, coalesced) ---
    {
        float acc = 0.f;
        for (int m = 0; m < LOTHER; ++m) acc += other[m * DDIM + tid];
        sVec[tid] = acc * (evidSide ? (1.0f / 256.f) : (1.0f / 512.f));
    }
    if (tid < ODIM) sw[tid] = wv[tid];
    __syncthreads();
    // --- pbias[n] = pooled @ W[D:, n] ---
    if (tid < ODIM) {
        float acc = 0.f;
        for (int k = 0; k < DDIM; ++k) acc += sVec[k] * W[(DDIM + k) * ODIM + tid];
        sPbias[tid] = acc;
    }
    __syncthreads();

    // --- WMMA GEMM: (L x 256) x (256 x 128), bf16 inputs, f32 accum ---
    const int wid  = tid >> 5;
    const int lane = tid & 31;
    const int lnn  = lane & 15;
    const int hi   = lane >> 4;              // 0 = lanes 0-15, 1 = lanes 16-31
    const int tilesPerWave = L >> 7;         // 4 (L=512) or 2 (L=256)

    // B fragment base for this lane: col n = nt*16 + lnn, K offset hi*16
    const unsigned short* bcol = sW + lnn * DDIM + hi * 16;

    #pragma unroll 1
    for (int t = 0; t < tilesPerWave; ++t) {
        const int mt = wid * tilesPerWave + t;
        v8f acc[8];
        #pragma unroll
        for (int nt = 0; nt < 8; ++nt) {
            const float b = sPbias[nt * 16 + lnn];     // bias per N column
            #pragma unroll
            for (int j = 0; j < 8; ++j) acc[nt][j] = b;
        }

        // A row base for this lane (lanes 0-15: K=k0..15; 16-31: K=k0+16..31)
        const float* arow = seq + (mt * 16 + lnn) * DDIM + hi * 16;

        // prefetch K-step 0
        float4 f0 = ((const float4*)arow)[0];
        float4 f1 = ((const float4*)arow)[1];
        float4 f2 = ((const float4*)(arow + 8))[0];
        float4 f3 = ((const float4*)(arow + 8))[1];

        #pragma unroll 1
        for (int ks = 0; ks < 8; ++ks) {
            // pack current A fragment
            v8u au;
            au[0] = pack2bf(f0.x, f0.y); au[1] = pack2bf(f0.z, f0.w);
            au[2] = pack2bf(f1.x, f1.y); au[3] = pack2bf(f1.z, f1.w);
            au[4] = pack2bf(f2.x, f2.y); au[5] = pack2bf(f2.z, f2.w);
            au[6] = pack2bf(f3.x, f3.y); au[7] = pack2bf(f3.z, f3.w);
            const v16bf A = __builtin_bit_cast(v16bf, au);

            // issue prefetch for next K step (wraps harmlessly at ks=7)
            const float* ap = arow + (((ks + 1) & 7) * 32);
            f0 = ((const float4*)ap)[0];
            f1 = ((const float4*)ap)[1];
            f2 = ((const float4*)(ap + 8))[0];
            f3 = ((const float4*)(ap + 8))[1];

            const unsigned short* bks = bcol + ks * 32;
            #pragma unroll
            for (int nt = 0; nt < 8; ++nt) {
                const unsigned short* bp = bks + nt * (16 * DDIM);
                const uint4 x0 = ((const uint4*)bp)[0];
                const uint4 x1 = ((const uint4*)bp)[1];
                const v8u bu = {x0.x, x0.y, x0.z, x0.w, x1.x, x1.y, x1.z, x1.w};
                const v16bf Bm = __builtin_bit_cast(v16bf, bu);
                acc[nt] = __builtin_amdgcn_wmma_f32_16x16x32_bf16(
                    false, A, false, Bm, (short)0, acc[nt], false, false);
            }
        }
        // --- epilogue: att[m] = sum_n tanh(h[m,n]) * w[n] ---
        float p[8];
        #pragma unroll
        for (int j = 0; j < 8; ++j) p[j] = 0.f;
        #pragma unroll
        for (int nt = 0; nt < 8; ++nt) {
            const float wn = sw[nt * 16 + lnn];
            #pragma unroll
            for (int j = 0; j < 8; ++j) p[j] += fast_tanh(acc[nt][j]) * wn;
        }
        #pragma unroll
        for (int j = 0; j < 8; ++j) {           // reduce over the 16 N-lanes
            #pragma unroll
            for (int off = 1; off < 16; off <<= 1)
                p[j] += __shfl_xor(p[j], off, 32);
        }
        if (lnn == 0) {
            #pragma unroll
            for (int j = 0; j < 8; ++j) {
                const int m = mt * 16 + j + hi * 8;   // C layout: VGPR j -> M=j / j+8
                sAtt[m] = mask[m] ? p[j] : NEGV;
            }
        }
    }
    __syncthreads();

    // --- masked softmax over L ---
    float lmax = -3.0e38f;
    for (int m = tid; m < L; m += 256) lmax = fmaxf(lmax, sAtt[m]);
    sVec[tid] = lmax;
    __syncthreads();
    for (int s = 128; s > 0; s >>= 1) {
        if (tid < s) sVec[tid] = fmaxf(sVec[tid], sVec[tid + s]);
        __syncthreads();
    }
    const float gmax = sVec[0];
    __syncthreads();
    float lsum = 0.f;
    for (int m = tid; m < L; m += 256) {
        const float e = __expf(sAtt[m] - gmax);
        sAtt[m] = e;
        lsum += e;
    }
    sVec[tid] = lsum;
    __syncthreads();
    for (int s = 128; s > 0; s >>= 1) {
        if (tid < s) sVec[tid] += sVec[tid + s];
        __syncthreads();
    }
    const float inv = 1.0f / sVec[0];

    // --- out[d] = sum_m score[m] * seq[m,d]  (thread t = dim d, coalesced rows) ---
    float acc = 0.f;
    for (int m = 0; m < L; ++m) acc += sAtt[m] * seq[m * DDIM + tid];
    out[tid] = acc * inv;
}

extern "C" void kernel_launch(void* const* d_in, const int* in_sizes, int n_in,
                              void* d_out, int out_size, void* d_ws, size_t ws_size,
                              hipStream_t stream) {
    (void)in_sizes; (void)n_in; (void)out_size; (void)d_ws; (void)ws_size;
    const float* claim = (const float*)d_in[0];
    const int*   cmask = (const int*)d_in[1];
    const float* evid  = (const float*)d_in[2];
    const int*   emask = (const int*)d_in[3];
    const float* W1    = (const float*)d_in[4];
    const float* w2    = (const float*)d_in[5];
    const float* W2    = (const float*)d_in[6];
    const float* w1    = (const float*)d_in[7];

    float* c_hat = (float*)d_out;                     // (32,8,1,256)
    float* e_hat = (float*)d_out + 32 * 8 * 256;      // (32,8,1,256)

    // Single fused launch: blocks 0..255 evidence side, 256..511 claim side.
    coattend_kernel<<<dim3(512), dim3(256), 0, stream>>>(
        claim, cmask, evid, emask, W1, w2, W2, w1, c_hat, e_hat);
}
